// MultiHeadSelfAttention_37048387895402
// MI455X (gfx1250) — compile-verified, matrix operand order checked by reading, not run on hardware
//
#include <hip/hip_runtime.h>

// ---------------- types ----------------
typedef __attribute__((ext_vector_type(16))) __bf16 v16bf;
typedef __attribute__((ext_vector_type(8)))  __bf16 v8bf;
typedef __attribute__((ext_vector_type(8)))  float  v8f;
typedef __attribute__((ext_vector_type(4)))  int    v4i;

#define B_  2
#define L_  2048
#define D_  1024
#define H_  16
#define HD_ 64
#define R_  17   // 2*MAXREL+1

// CDNA5 async global->LDS path (guarded: falls back to direct global loads)
#if defined(__gfx1250__) && __has_builtin(__builtin_amdgcn_global_load_async_to_lds_b128)
#define USE_ASYNC_LDS 1
#else
#define USE_ASYNC_LDS 0
#endif

template <int N>
__device__ __forceinline__ void wait_async() {
#if defined(__gfx1250__)
#if __has_builtin(__builtin_amdgcn_s_wait_asynccnt)
  __builtin_amdgcn_s_wait_asynccnt((short)N);
#else
  asm volatile("s_wait_asynccnt %0" ::"i"(N) : "memory");
#endif
#endif
}

// ---------------- WMMA helpers ----------------
__device__ __forceinline__ v8f wmma_bf16(v16bf a, v16bf b, v8f c) {
  return __builtin_amdgcn_wmma_f32_16x16x32_bf16(
      /*neg_a=*/false, a, /*neg_b=*/false, b,
      /*c_mod=*/(short)0, c, /*reuse_a=*/false, /*reuse_b=*/false);
}

// A/B fragment for v_wmma_*_16x16x32 (16-bit): lane m%16 holds row (row0 + m%16);
// lanes 0-15 hold K = {k0..k0+7, k0+16..k0+23}, lanes 16-31 hold K = {k0+8..k0+15, k0+24..k0+31}.
__device__ __forceinline__ v16bf load_frag_bf16(const __bf16* base,
                                                int row0, int ld, int k0, int lane) {
  int r  = row0 + (lane & 15);
  int kh = (lane >> 4) << 3;
  const __bf16* p = base + (size_t)r * ld + k0 + kh;
  union { v16bf v; v8bf h[2]; } u;
  u.h[0] = *(const v8bf*)(p);
  u.h[1] = *(const v8bf*)(p + 16);
  return u.v;
}

// Same fragment but sourced from fp32 (global or LDS), converted to bf16 in-register.
__device__ __forceinline__ v16bf load_frag_f32(const float* base,
                                               int row0, int ld, int k0, int lane) {
  int r  = row0 + (lane & 15);
  int kh = (lane >> 4) << 3;
  const float* p = base + (size_t)r * ld + k0 + kh;
  union { v16bf v; __bf16 e[16]; } u;
#pragma unroll
  for (int j = 0; j < 8; ++j) {
    u.e[j]     = (__bf16)p[j];
    u.e[8 + j] = (__bf16)p[16 + j];
  }
  return u.v;
}

// ---------------- Kernel 1: packed QKV projection ----------------
// qkv[n,e] = sum_d x[n,d] * w_in[e,d];  n in [0,4096), e in [0,3072)
// q -> qb[B,H,L,HD] (scaled 1/8), k -> kb[B,H,L,HD], v -> vt[B,H,HD,L] (transposed)
// Each wave computes a 16x64 strip (4 N-tiles) so the A fragment feeds 4 WMMAs.
__global__ void __launch_bounds__(128)
qkv_kernel(const float* __restrict__ x, const float* __restrict__ w_in,
           __bf16* __restrict__ qb, __bf16* __restrict__ kb, __bf16* __restrict__ vt) {
  int lane = threadIdx.x & 31;
  int wave = threadIdx.x >> 5;
  int mt = blockIdx.x;              // 256 row tiles
  int ng = blockIdx.y * 4 + wave;   // 48 groups of 4 col tiles
  v8f c[4] = {};
  for (int k0 = 0; k0 < D_; k0 += 32) {
    v16bf a = load_frag_f32(x, mt * 16, D_, k0, lane);
    if (k0 + 256 < D_)
      __builtin_prefetch(w_in + (size_t)(ng * 64 + (lane & 15)) * D_ + k0 + 256, 0, 0);
#pragma unroll
    for (int j = 0; j < 4; ++j) {
      v16bf b = load_frag_f32(w_in, (ng * 4 + j) * 16, D_, k0, lane);
      c[j] = wmma_bf16(a, b, c[j]);
    }
  }
  // region is wave-uniform: ng 0-15 -> q, 16-31 -> k, 32-47 -> v
  int region = ng >> 4;
  int n  = lane & 15;
  int mb = (lane >> 4) << 3;
  int nrow0 = mt * 16 + mb;                 // 8 consecutive output rows per lane
  int bb = nrow0 >> 11, l0 = nrow0 & (L_ - 1);
  if (region == 2) {
    // V transposed: 8 consecutive l at fixed (h,d) -> one 16-byte store per tile
#pragma unroll
    for (int j = 0; j < 4; ++j) {
      int e = (ng * 4 + j) * 16 + n - 2048;
      int h = e >> 6, d = e & 63;
      union { v8bf v; __bf16 e8[8]; } u;
#pragma unroll
      for (int r = 0; r < 8; ++r) u.e8[r] = (__bf16)c[j][r];
      *(v8bf*)(vt + (((size_t)bb * H_ + h) * HD_ + d) * L_ + l0) = u.v;
    }
  } else {
    __bf16* dst = (region == 0) ? qb : kb;
    float scale = (region == 0) ? 0.125f : 1.0f;
    int ebase = region * D_;
#pragma unroll
    for (int j = 0; j < 4; ++j) {
      int e = (ng * 4 + j) * 16 + n - ebase;
      int h = e >> 6, d = e & 63;
      __bf16* p = dst + (((size_t)bb * H_ + h) * L_ + l0) * HD_ + d;
#pragma unroll
      for (int r = 0; r < 8; ++r) p[(size_t)r * HD_] = (__bf16)(c[j][r] * scale);
    }
  }
}

// ---------------- Kernel 2: fused scores + softmax + PV + rel terms ----------------
struct Smem {
  float  sc[16][L_];          // 128 KB fp32 score strip
  __bf16 pb[16][L_];          //  64 KB bf16 probs (for PV WMMA A-fragments)
  __bf16 kst[8][2][16 * HD_]; //  32 KB per-wave double-buffered K tiles (async staging)
  float  qrel[16][R_];        // q . rel_k_emb[r]
  float  pw[16][R_];          // bucketed prob sums for rel_v
  float  red[16][16];         // softmax reduction scratch
  float  outp[8][16][HD_];    //  32 KB per-wave PV partials
};
extern __shared__ char smem_raw[];

__device__ __forceinline__ void stage_ktile(const __bf16* kbase, int kt,
                                            __bf16* dst, int lane) {
#if USE_ASYNC_LDS
  // one K tile = 16 rows x 64 bf16 = 2048 B = 4 x (32 lanes x 16 B)
#pragma unroll
  for (int j = 0; j < 4; ++j) {
    int chunk = j * 32 + lane;
    const __bf16* gp = kbase + (size_t)kt * (16 * HD_) + chunk * 8;
    __bf16* lp = dst + chunk * 8;
    __builtin_amdgcn_global_load_async_to_lds_b128(
        (__attribute__((address_space(1))) v4i*)(void*)(__bf16*)gp,
        (__attribute__((address_space(3))) v4i*)(void*)lp, 0, 0);
  }
#else
  (void)kbase; (void)kt; (void)dst; (void)lane;
#endif
}

__global__ void __launch_bounds__(256)
attn_kernel(const __bf16* __restrict__ qb, const __bf16* __restrict__ kb,
            const __bf16* __restrict__ vt,
            const float* __restrict__ rel_k_emb, const float* __restrict__ rel_v_emb,
            const float* __restrict__ rel_bias,
            float* __restrict__ probs, __bf16* __restrict__ ao) {
  Smem& s = *(Smem*)smem_raw;
  int tid = threadIdx.x, lane = tid & 31, wave = tid >> 5;   // 8 waves
  int qt = blockIdx.x;   // 128 query tiles
  int h  = blockIdx.y;   // 16 heads
  int b  = blockIdx.z;   // 2 batches
  const __bf16* qbase = qb + ((size_t)b * H_ + h) * L_ * HD_;
  const __bf16* kbase = kb + ((size_t)b * H_ + h) * L_ * HD_;
  const __bf16* vbase = vt + ((size_t)b * H_ + h) * HD_ * L_;
  float bias = rel_bias[h];

  // phase 0: qrel[q][r] = q_vec . rel_k_emb[r]; zero pw
  for (int i = tid; i < 16 * R_; i += 256) {
    int qr = i / R_, r = i % R_;
    const __bf16* qv = qbase + (size_t)(qt * 16 + qr) * HD_;
    float acc = 0.f;
    for (int d = 0; d < HD_; ++d) acc += (float)qv[d] * rel_k_emb[r * HD_ + d];
    s.qrel[qr][r] = acc;
    s.pw[qr][r] = 0.f;
  }
  __syncthreads();

  // phase 1: scores = Q K^T + qrel(bucket) + bias   (each wave owns 16 k-tiles)
  v16bf qa0 = load_frag_bf16(qbase, qt * 16, HD_, 0, lane);
  v16bf qa1 = load_frag_bf16(qbase, qt * 16, HD_, 32, lane);
#if USE_ASYNC_LDS
  __bf16* buf0 = &s.kst[wave][0][0];
  __bf16* buf1 = &s.kst[wave][1][0];
  stage_ktile(kbase, wave * 16, buf0, lane);
#endif
  for (int lt = 0; lt < 16; ++lt) {
    int kt = wave * 16 + lt;
    v16bf kf0, kf1;
#if USE_ASYNC_LDS
    if (lt + 1 < 16) {
      stage_ktile(kbase, kt + 1, (lt & 1) ? buf0 : buf1, lane);
      wait_async<4>();            // previous tile's 4 async loads complete (in-order)
    } else {
      wait_async<0>();
    }
    const __bf16* kt_lds = (lt & 1) ? buf1 : buf0;
    kf0 = load_frag_bf16(kt_lds, 0, HD_, 0, lane);
    kf1 = load_frag_bf16(kt_lds, 0, HD_, 32, lane);
#else
    kf0 = load_frag_bf16(kbase, kt * 16, HD_, 0, lane);
    kf1 = load_frag_bf16(kbase, kt * 16, HD_, 32, lane);
#endif
    v8f c = {};
    c = wmma_bf16(qa0, kf0, c);
    c = wmma_bf16(qa1, kf1, c);
    int kg = kt * 16 + (lane & 15);
#pragma unroll
    for (int r = 0; r < 8; ++r) {
      int m = r + ((lane >> 4) << 3);
      int qg = qt * 16 + m;
      int dd = kg - qg;
      int u = dd < -8 ? 0 : (dd > 8 ? 16 : dd + 8);
      s.sc[m][kg] = c[r] + s.qrel[m][u] + bias;
    }
  }
  __syncthreads();

  // phase 2: softmax over L (16 threads per row, 128 cols each)
  int row = tid >> 4, seg = tid & 15, c0 = seg * 128;
  float mx = -3.4e38f;
  for (int cc = 0; cc < 128; ++cc) mx = fmaxf(mx, s.sc[row][c0 + cc]);
  s.red[row][seg] = mx;
  __syncthreads();
  float rmax = s.red[row][0];
  for (int j = 1; j < 16; ++j) rmax = fmaxf(rmax, s.red[row][j]);
  __syncthreads();
  float sum = 0.f;
  for (int cc = 0; cc < 128; ++cc) {
    float e = __expf(s.sc[row][c0 + cc] - rmax);
    s.sc[row][c0 + cc] = e;
    sum += e;
  }
  s.red[row][seg] = sum;
  __syncthreads();
  float rsum = 0.f;
  for (int j = 0; j < 16; ++j) rsum += s.red[row][j];
  float inv = 1.0f / rsum;
  int qg = qt * 16 + row;
  float* prow = probs + (((size_t)b * H_ + h) * L_ + qg) * L_;
  float pwloc[R_];
#pragma unroll
  for (int r = 0; r < R_; ++r) pwloc[r] = 0.f;
  for (int cc = 0; cc < 128; ++cc) {
    int kg = c0 + cc;
    float p = s.sc[row][kg] * inv;
    prow[kg] = p;                  // fp32 probs output (written once)
    s.pb[row][kg] = (__bf16)p;     // bf16 copy for PV WMMA
    int dd = kg - qg;
    int u = dd < -8 ? 0 : (dd > 8 ? 16 : dd + 8);
    pwloc[u] += p;
  }
#pragma unroll
  for (int r = 0; r < R_; ++r)
    if (pwloc[r] != 0.f) atomicAdd(&s.pw[row][r], pwloc[r]);
  __syncthreads();

  // phase 3: PV (each wave owns a 256-wide K slice), A-fragments from bf16 LDS
  for (int ntl = 0; ntl < 4; ++ntl) {
    v8f acc = {};
    for (int ks = 0; ks < 8; ++ks) {
      int k0 = wave * 256 + ks * 32;
      v16bf a  = load_frag_bf16(&s.pb[0][0], 0, L_, k0, lane);   // probs
      v16bf bf = load_frag_bf16(vbase, ntl * 16, L_, k0, lane);  // V^T rows = d
      acc = wmma_bf16(a, bf, acc);
    }
    int n = lane & 15;
#pragma unroll
    for (int r = 0; r < 8; ++r) {
      int m = r + ((lane >> 4) << 3);
      s.outp[wave][m][ntl * 16 + n] = acc[r];
    }
  }
  __syncthreads();

  // phase 4: reduce wave partials + pw @ rel_v_emb, write bf16 attn-out [B,L,D]
  for (int i = tid; i < 16 * HD_; i += 256) {
    int qr = i >> 6, d = i & 63;
    float v = 0.f;
#pragma unroll
    for (int w = 0; w < 8; ++w) v += s.outp[w][qr][d];
#pragma unroll
    for (int r = 0; r < R_; ++r) v += s.pw[qr][r] * rel_v_emb[r * HD_ + d];
    int lg = qt * 16 + qr;
    ao[((size_t)b * L_ + lg) * D_ + h * HD_ + d] = (__bf16)v;
  }
}

// ---------------- Kernel 3: output projection ----------------
// out[n,d] = sum_e ao[n,e] * w_out[d,e]; each wave computes 16x64 (4 N-tiles)
__global__ void __launch_bounds__(128)
proj_kernel(const __bf16* __restrict__ ao, const float* __restrict__ w_out,
            float* __restrict__ out) {
  int lane = threadIdx.x & 31;
  int wave = threadIdx.x >> 5;
  int mt = blockIdx.x;              // 256 row tiles
  int ng = blockIdx.y * 4 + wave;   // 16 groups of 4 col tiles
  v8f c[4] = {};
  for (int k0 = 0; k0 < D_; k0 += 32) {
    v16bf a = load_frag_bf16(ao, mt * 16, D_, k0, lane);
    if (k0 + 256 < D_)
      __builtin_prefetch(w_out + (size_t)(ng * 64 + (lane & 15)) * D_ + k0 + 256, 0, 0);
#pragma unroll
    for (int j = 0; j < 4; ++j) {
      v16bf b = load_frag_f32(w_out, (ng * 4 + j) * 16, D_, k0, lane);
      c[j] = wmma_bf16(a, b, c[j]);
    }
  }
  int mbase = mt * 16 + ((lane >> 4) << 3);
#pragma unroll
  for (int j = 0; j < 4; ++j) {
    int d = (ng * 4 + j) * 16 + (lane & 15);
#pragma unroll
    for (int r = 0; r < 8; ++r) out[(size_t)(mbase + r) * D_ + d] = c[j][r];
  }
}

// ---------------- launch ----------------
extern "C" void kernel_launch(void* const* d_in, const int* in_sizes, int n_in,
                              void* d_out, int out_size, void* d_ws, size_t ws_size,
                              hipStream_t stream) {
  const float* x         = (const float*)d_in[0];
  const float* w_in      = (const float*)d_in[1];
  const float* w_out     = (const float*)d_in[2];
  const float* rel_k_emb = (const float*)d_in[3];
  const float* rel_v_emb = (const float*)d_in[4];
  const float* rel_bias  = (const float*)d_in[5];

  const size_t probs_elems = (size_t)B_ * H_ * L_ * L_;   // 134,217,728
  float* probs = (float*)d_out;
  float* out   = (float*)d_out + probs_elems;

  char* ws = (char*)d_ws;
  const size_t buf_bytes = (size_t)B_ * H_ * L_ * HD_ * sizeof(__bf16);  // 8 MB each
  __bf16* qb = (__bf16*)(ws + 0 * buf_bytes);
  __bf16* kb = (__bf16*)(ws + 1 * buf_bytes);
  __bf16* vt = (__bf16*)(ws + 2 * buf_bytes);
  __bf16* ao = (__bf16*)(ws + 3 * buf_bytes);

  // Kernel 1: QKV projection (256 x 48 wave-strips of 16x64)
  qkv_kernel<<<dim3(256, 12), 128, 0, stream>>>(x, w_in, qb, kb, vt);

  // Kernel 2: fused attention; ~260 KB dynamic LDS (320 KB/WGP available)
  attn_kernel<<<dim3(L_ / 16, H_, B_), 256, sizeof(Smem), stream>>>(
      qb, kb, vt, rel_k_emb, rel_v_emb, rel_bias, probs, ao);

  // Kernel 3: output projection (256 x 16 wave-strips of 16x64)
  proj_kernel<<<dim3(256, 4), 128, 0, stream>>>(ao, w_out, out);
}
